// AttentionBlock_31739808317922
// MI455X (gfx1250) — matrix operand order repaired and checked by reading, not compiled, not run on hardware
//
#include <hip/hip_runtime.h>
#include <hip/hip_bf16.h>

// ---------------------------------------------------------------------------
// AttentionBlock for MI455X (gfx1250): GroupNorm + QKV + flash-attention + proj
// All GEMM/attention math runs on V_WMMA_F32_16X16X32_F16 (wave32).
// Round 2: software-pipelined GEMMs (double-buffered fragments), attention
// load/compute overlap, global_prefetch_b8 for streamed K/V tiles.
// ---------------------------------------------------------------------------

#define B_   2
#define C_   512
#define N_   4096        // 64*64 tokens
#define HH_  8           // heads
#define D_   64          // head dim
#define G_   32          // groupnorm groups

typedef __attribute__((ext_vector_type(16))) _Float16 v16h;
typedef __attribute__((ext_vector_type(8)))  _Float16 v8h;
typedef __attribute__((ext_vector_type(8)))  float    v8f;

// Static device scratch (fully rewritten every call; independent of ws_size).
__device__ _Float16 g_xh  [B_*N_*C_];        // normalized x, [B][N][C] f16
__device__ _Float16 g_q   [B_*HH_*N_*D_];    // [B][H][N][D]
__device__ _Float16 g_k   [B_*HH_*N_*D_];    // [B][H][N][D]
__device__ _Float16 g_vT  [B_*HH_*D_*N_];    // [B][H][D][N]  (transposed for P@V)
__device__ _Float16 g_attn[B_*N_*C_];        // attention out, [B][N][C] f16
__device__ _Float16 g_wq[C_*C_];
__device__ _Float16 g_wk[C_*C_];
__device__ _Float16 g_wv[C_*C_];
__device__ _Float16 g_wp[C_*C_];

__device__ __forceinline__ v8f v8f_zero() {
    v8f z;
#pragma unroll
    for (int i = 0; i < 8; ++i) z[i] = 0.0f;
    return z;
}

// A-fragment (16x32 f16, MxK): lane holds row (lane&15); K = {ko..ko+7, ko+16..ko+23},
// ko = 8*(lane>=16).  `base` points at element (row0=0, k0=0); row-major, K contiguous.
__device__ __forceinline__ v16h load_frag_a(const _Float16* base, int ld, int lane) {
    const int r  = lane & 15;
    const int ko = (lane >> 4) << 3;
    const _Float16* p = base + r * ld + ko;
    v8h lo = *(const v8h*)(p);
    v8h hi = *(const v8h*)(p + 16);
    return __builtin_shufflevector(lo, hi, 0,1,2,3,4,5,6,7,8,9,10,11,12,13,14,15);
}

// B-fragment (32x16 f16, KxN): lane holds column (lane&15); K = {ko..ko+7, ko+16..ko+23},
// ko = 8*(lane>=16).  Element (k,n) is stored at base[n*ld + k].
__device__ __forceinline__ v16h load_frag_b(const _Float16* base, int ld, int lane) {
    const int c  = lane & 15;
    const int kb = (lane >> 4) << 3;
    const _Float16* p = base + c * ld + kb;
    v8h lo = *(const v8h*)(p);
    v8h hi = *(const v8h*)(p + 16);
    return __builtin_shufflevector(lo, hi, 0,1,2,3,4,5,6,7,8,9,10,11,12,13,14,15);
}

__device__ __forceinline__ v8f wmma_f16(v16h a, v16h b, v8f c) {
    return __builtin_amdgcn_wmma_f32_16x16x32_f16(false, a, false, b, (short)0, c,
                                                  false, false);
}

// ---------------------------------------------------------------------------
// Kernel 0: fp32 -> fp16 weight conversion (all four 512x512 matrices)
// ---------------------------------------------------------------------------
__global__ void cvt_weights_kernel(const float* __restrict__ qw,
                                   const float* __restrict__ kw,
                                   const float* __restrict__ vw,
                                   const float* __restrict__ pw) {
    const int i = blockIdx.x * 256 + threadIdx.x;   // 0 .. 262143
    g_wq[i] = (_Float16)qw[i];
    g_wk[i] = (_Float16)kw[i];
    g_wv[i] = (_Float16)vw[i];
    g_wp[i] = (_Float16)pw[i];
}

// ---------------------------------------------------------------------------
// Kernel 1: GroupNorm over (C/G=16 channels x 4096 spatial), write f16 [B][N][C]
// One block per (b, g); 256 threads.
// ---------------------------------------------------------------------------
__global__ void groupnorm_kernel(const float* __restrict__ x,
                                 const float* __restrict__ w,
                                 const float* __restrict__ bias) {
    const int b = blockIdx.x >> 5;
    const int g = blockIdx.x & 31;
    const int tid = threadIdx.x;
    const float* xg = x + ((size_t)b * C_ + g * 16) * N_;

    float s = 0.0f, ss = 0.0f;
    for (int i = tid; i < 16 * N_; i += 256) {          // coalesced over n
        const float v = xg[i];
        s += v; ss += v * v;
    }
    __shared__ float rs[256];
    __shared__ float rq[256];
    rs[tid] = s; rq[tid] = ss;
    __syncthreads();
#pragma unroll
    for (int st = 128; st > 0; st >>= 1) {
        if (tid < st) { rs[tid] += rs[tid + st]; rq[tid] += rq[tid + st]; }
        __syncthreads();
    }
    const float inv_cnt = 1.0f / (16.0f * N_);
    const float mean = rs[0] * inv_cnt;
    const float var  = rq[0] * inv_cnt - mean * mean;
    const float rstd = rsqrtf(var + 1e-5f);

    // second pass: channel-fast iteration so f16 writes to [n][c] are contiguous
    for (int i = tid; i < 16 * N_; i += 256) {
        const int cl = i & 15;
        const int n  = i >> 4;
        const int c  = g * 16 + cl;
        const float v = (xg[cl * N_ + n] - mean) * rstd * w[c] + bias[c];
        g_xh[((size_t)b * N_ + n) * C_ + c] = (_Float16)v;
    }
}

// ---------------------------------------------------------------------------
// Kernel 2: QKV projection GEMM, double-buffered over K so global loads for
// step k+32 are in flight while the WMMAs for step k execute.
// Block = 8 waves; tile 128 tokens x 64 cos; K = 512.
// sel: 0 -> Q ([B][H][N][D]), 1 -> K ([B][H][N][D]), 2 -> V transposed ([B][H][D][N])
// ---------------------------------------------------------------------------
__global__ void gemm_qkv_kernel(const float* __restrict__ bias, int sel) {
    const int lane = threadIdx.x & 31;
    const int wave = threadIdx.x >> 5;
    const int row0 = blockIdx.x * 128 + wave * 16;   // token tile (over B*N = 8192)
    const int col0 = blockIdx.y * 64;                // output-channel tile

    const _Float16* wm = (sel == 0) ? g_wq : (sel == 1) ? g_wk : g_wv;
    const _Float16* xb = g_xh + (size_t)row0 * C_;

    v8f acc[4];
#pragma unroll
    for (int t = 0; t < 4; ++t) acc[t] = v8f_zero();

    // prologue: fragments for k = 0
    v16h ac = load_frag_a(xb, C_, lane);
    v16h bc[4];
#pragma unroll
    for (int t = 0; t < 4; ++t)
        bc[t] = load_frag_b(wm + (size_t)(col0 + t * 16) * C_, C_, lane);

    for (int k = 32; k < C_; k += 32) {
        // issue next-step loads BEFORE consuming current fragments
        v16h an = load_frag_a(xb + k, C_, lane);
        v16h bn[4];
#pragma unroll
        for (int t = 0; t < 4; ++t)
            bn[t] = load_frag_b(wm + (size_t)(col0 + t * 16) * C_ + k, C_, lane);
#pragma unroll
        for (int t = 0; t < 4; ++t) acc[t] = wmma_f16(ac, bc[t], acc[t]);
        ac = an;
#pragma unroll
        for (int t = 0; t < 4; ++t) bc[t] = bn[t];
    }
#pragma unroll
    for (int t = 0; t < 4; ++t) acc[t] = wmma_f16(ac, bc[t], acc[t]);

    const int rlo = (lane >> 4) << 3;   // 0 or 8 (M half)
    const int cL  = lane & 15;
#pragma unroll
    for (int t = 0; t < 4; ++t) {
        const int co = col0 + t * 16 + cL;
        const float bv = bias[co];
        const int h = co >> 6, d = co & 63;
#pragma unroll
        for (int i = 0; i < 8; ++i) {
            const int tok = row0 + rlo + i;
            const int b = tok >> 12, n = tok & (N_ - 1);
            const float val = acc[t][i] + bv;
            if (sel == 0)
                g_q[(((size_t)b * HH_ + h) * N_ + n) * D_ + d] = (_Float16)val;
            else if (sel == 1)
                g_k[(((size_t)b * HH_ + h) * N_ + n) * D_ + d] = (_Float16)val;
            else
                g_vT[(((size_t)b * HH_ + h) * D_ + d) * N_ + n] = (_Float16)val;
        }
    }
}

// ---------------------------------------------------------------------------
// Kernel 3: flash attention.  Grid (N/128, B*H); block = 8 waves.
// Each wave owns 16 query rows.  Per 64-key tile:
//   - prefetch next tile's K/V rows (global_prefetch_b8)
//   - load 8 K fragments, 8 score WMMAs
//   - issue 8 V fragment loads, then run the softmax VALU on top of them
//   - stage P through per-wave LDS (same-wave DS ordering), 8 P@V WMMAs
// ---------------------------------------------------------------------------
__global__ void attention_kernel() {
    const int lane = threadIdx.x & 31;
    const int wave = threadIdx.x >> 5;
    const int bh   = blockIdx.y;                      // b*8 + h
    const int q0   = blockIdx.x * 128 + wave * 16;    // query row tile

    const _Float16* Qb = g_q  + (size_t)bh * N_ * D_;
    const _Float16* Kb = g_k  + (size_t)bh * N_ * D_;
    const _Float16* Vb = g_vT + (size_t)bh * D_ * N_;

    __shared__ __align__(16) _Float16 lds[8][16 * 64];
    _Float16* myP = &lds[wave][0];

    // Q fragments for this wave's 16 rows (K-dim = 64 -> two 16x32 frags), kept live.
    const v16h qf0 = load_frag_a(Qb + (size_t)q0 * D_ + 0,  D_, lane);
    const v16h qf1 = load_frag_a(Qb + (size_t)q0 * D_ + 32, D_, lane);

    v8f o[4];
#pragma unroll
    for (int t = 0; t < 4; ++t) o[t] = v8f_zero();
    v8f m, l;
#pragma unroll
    for (int i = 0; i < 8; ++i) { m[i] = -3.0e38f; l[i] = 0.0f; }

    const float scale = 0.125f;                       // 1/sqrt(64)
    const int rlo = (lane >> 4) << 3;
    const int cL  = lane & 15;

    for (int kt = 0; kt < N_; kt += 64) {             // 64-key tiles
        // ---- prefetch next key tile (K rows + V rows), one line per lane ----
        if (kt + 64 < N_) {
            __builtin_prefetch(Kb + (size_t)(kt + 64 + lane) * D_, 0, 3);
            __builtin_prefetch(Kb + (size_t)(kt + 96 + lane) * D_, 0, 3);
            __builtin_prefetch(Vb + (size_t)(lane)      * N_ + kt + 64, 0, 3);
            __builtin_prefetch(Vb + (size_t)(lane + 32) * N_ + kt + 64, 0, 3);
        }

        // ---- load all 8 K fragments, then S = scale * Q K^T  (16 x 64) ----
        v16h kf[8];
#pragma unroll
        for (int t = 0; t < 4; ++t) {
            const _Float16* kb = Kb + (size_t)(kt + t * 16) * D_;
            kf[2 * t]     = load_frag_b(kb + 0,  D_, lane);
            kf[2 * t + 1] = load_frag_b(kb + 32, D_, lane);
        }
        v8f s[4];
#pragma unroll
        for (int t = 0; t < 4; ++t) {
            s[t] = v8f_zero();
            s[t] = wmma_f16(qf0, kf[2 * t],     s[t]);
            s[t] = wmma_f16(qf1, kf[2 * t + 1], s[t]);
        }

        // ---- issue V fragment loads now; softmax VALU below hides them ----
        v16h vf[8];
#pragma unroll
        for (int t = 0; t < 4; ++t) {
            const _Float16* vb = Vb + (size_t)(t * 16) * N_ + kt;
            vf[2 * t]     = load_frag_b(vb + 0,  N_, lane);
            vf[2 * t + 1] = load_frag_b(vb + 32, N_, lane);
        }

        // ---- online softmax: row max over 4 tiles, then over 16-lane half ----
#pragma unroll
        for (int t = 0; t < 4; ++t)
#pragma unroll
            for (int i = 0; i < 8; ++i) s[t][i] *= scale;

        v8f mx;
#pragma unroll
        for (int i = 0; i < 8; ++i)
            mx[i] = fmaxf(fmaxf(s[0][i], s[1][i]), fmaxf(s[2][i], s[3][i]));
#pragma unroll
        for (int off = 8; off > 0; off >>= 1)
#pragma unroll
            for (int i = 0; i < 8; ++i)
                mx[i] = fmaxf(mx[i], __shfl_xor(mx[i], off, 32));

        v8f corr;
#pragma unroll
        for (int i = 0; i < 8; ++i) {
            const float mn = fmaxf(m[i], mx[i]);
            corr[i] = __expf(m[i] - mn);
            m[i] = mn;
        }
#pragma unroll
        for (int t = 0; t < 4; ++t)
#pragma unroll
            for (int i = 0; i < 8; ++i) s[t][i] = __expf(s[t][i] - m[i]);
#pragma unroll
        for (int i = 0; i < 8; ++i)
            l[i] = l[i] * corr[i] + s[0][i] + s[1][i] + s[2][i] + s[3][i];
#pragma unroll
        for (int t = 0; t < 4; ++t)
#pragma unroll
            for (int i = 0; i < 8; ++i) o[t][i] *= corr[i];

        // ---- P (C/D layout) -> LDS row-major 16x64 -> A fragments ----
        // Same-wave LDS ops execute in order: no barrier needed (per-wave buffer).
#pragma unroll
        for (int t = 0; t < 4; ++t)
#pragma unroll
            for (int i = 0; i < 8; ++i)
                myP[(rlo + i) * 64 + t * 16 + cL] = (_Float16)s[t][i];

        const v16h p0 = load_frag_a(myP + 0,  64, lane);
        const v16h p1 = load_frag_a(myP + 32, 64, lane);

        // ---- O += P V  (V fragments already resident) ----
#pragma unroll
        for (int t = 0; t < 4; ++t) {
            o[t] = wmma_f16(p0, vf[2 * t],     o[t]);
            o[t] = wmma_f16(p1, vf[2 * t + 1], o[t]);
        }
    }

    // ---- finalize: full row sum across the 16-lane half, normalize, store ----
#pragma unroll
    for (int off = 8; off > 0; off >>= 1)
#pragma unroll
        for (int i = 0; i < 8; ++i) l[i] += __shfl_xor(l[i], off, 32);

    const int b = bh >> 3, h = bh & 7;
#pragma unroll
    for (int t = 0; t < 4; ++t)
#pragma unroll
        for (int i = 0; i < 8; ++i) {
            const float val = o[t][i] / l[i];
            const int n = q0 + rlo + i;
            g_attn[((size_t)b * N_ + n) * C_ + h * D_ + t * 16 + cL] = (_Float16)val;
        }
}

// ---------------------------------------------------------------------------
// Kernel 4: output projection + bias + residual, f32 output in [B][C][H*W]
// (same double-buffered K pipeline as the QKV GEMM)
// ---------------------------------------------------------------------------
__global__ void gemm_proj_kernel(const float* __restrict__ bias,
                                 const float* __restrict__ resid,
                                 float* __restrict__ out) {
    const int lane = threadIdx.x & 31;
    const int wave = threadIdx.x >> 5;
    const int row0 = blockIdx.x * 128 + wave * 16;   // token tile
    const int col0 = blockIdx.y * 64;                // output-channel tile
    const _Float16* ab = g_attn + (size_t)row0 * C_;

    v8f acc[4];
#pragma unroll
    for (int t = 0; t < 4; ++t) acc[t] = v8f_zero();

    v16h ac = load_frag_a(ab, C_, lane);
    v16h bc[4];
#pragma unroll
    for (int t = 0; t < 4; ++t)
        bc[t] = load_frag_b(g_wp + (size_t)(col0 + t * 16) * C_, C_, lane);

    for (int k = 32; k < C_; k += 32) {
        v16h an = load_frag_a(ab + k, C_, lane);
        v16h bn[4];
#pragma unroll
        for (int t = 0; t < 4; ++t)
            bn[t] = load_frag_b(g_wp + (size_t)(col0 + t * 16) * C_ + k, C_, lane);
#pragma unroll
        for (int t = 0; t < 4; ++t) acc[t] = wmma_f16(ac, bc[t], acc[t]);
        ac = an;
#pragma unroll
        for (int t = 0; t < 4; ++t) bc[t] = bn[t];
    }
#pragma unroll
    for (int t = 0; t < 4; ++t) acc[t] = wmma_f16(ac, bc[t], acc[t]);

    const int rlo = (lane >> 4) << 3;
    const int cL  = lane & 15;
#pragma unroll
    for (int t = 0; t < 4; ++t) {
        const int co = col0 + t * 16 + cL;
        const float bv = bias[co];
#pragma unroll
        for (int i = 0; i < 8; ++i) {
            const int tok = row0 + rlo + i;
            const int b = tok >> 12, n = tok & (N_ - 1);
            const size_t idx = ((size_t)b * C_ + co) * N_ + n;
            out[idx] = acc[t][i] + bv + resid[idx];
        }
    }
}

// ---------------------------------------------------------------------------
extern "C" void kernel_launch(void* const* d_in, const int* in_sizes, int n_in,
                              void* d_out, int out_size, void* d_ws, size_t ws_size,
                              hipStream_t stream) {
    const float* hs     = (const float*)d_in[0];
    const float* gn_w   = (const float*)d_in[1];
    const float* gn_b   = (const float*)d_in[2];
    const float* q_w    = (const float*)d_in[3];
    const float* q_b    = (const float*)d_in[4];
    const float* k_w    = (const float*)d_in[5];
    const float* k_b    = (const float*)d_in[6];
    const float* v_w    = (const float*)d_in[7];
    const float* v_b    = (const float*)d_in[8];
    const float* proj_w = (const float*)d_in[9];
    const float* proj_b = (const float*)d_in[10];
    float* out = (float*)d_out;
    (void)in_sizes; (void)n_in; (void)out_size; (void)d_ws; (void)ws_size;

    cvt_weights_kernel<<<(C_ * C_) / 256, 256, 0, stream>>>(q_w, k_w, v_w, proj_w);
    groupnorm_kernel<<<B_ * G_, 256, 0, stream>>>(hs, gn_w, gn_b);

    dim3 ggrid((B_ * N_) / 128, C_ / 64);
    gemm_qkv_kernel<<<ggrid, 256, 0, stream>>>(q_b, 0);
    gemm_qkv_kernel<<<ggrid, 256, 0, stream>>>(k_b, 1);
    gemm_qkv_kernel<<<ggrid, 256, 0, stream>>>(v_b, 2);

    attention_kernel<<<dim3(N_ / 128, B_ * HH_), 256, 0, stream>>>();

    gemm_proj_kernel<<<ggrid, 256, 0, stream>>>(proj_b, hs, out);
}
